// Transformer_encoder_21938692948179
// MI455X (gfx1250) — matrix operand order repaired and checked by reading, not compiled, not run on hardware
//
#include <hip/hip_runtime.h>

// ---------------------------------------------------------------------------
// Graph-transformer block for MI455X (gfx1250, wave32, WMMA).
// Heavy E-path fused into one WMMA mega-kernel (async Global->LDS staging);
// Y stored f16 for online softmax; X / y / stats paths are tiny VALU kernels.
// ---------------------------------------------------------------------------

typedef __attribute__((ext_vector_type(16))) _Float16 v16h;
typedef __attribute__((ext_vector_type(8)))  float    v8f;

#define EPS 1e-5f

// ---------------- WMMA helpers -------------------------------------------

__device__ inline v8f wmma_f16(v16h a, v16h b, v8f c) {
  // D = A(16x32 f16) * B(32x16 f16) + C(16x16 f32)
  return __builtin_amdgcn_wmma_f32_16x16x32_f16(
      /*neg_a=*/false, a, /*neg_b=*/false, b,
      /*c_mod=*/(short)0, c, /*reuse_a=*/false, /*reuse_b=*/false);
}

// Gather an A fragment (16x32 slice, f32 source in LDS -> f16 regs).
// K-contiguous A order (matches pack_b16): lanes 0-15 hold M=lane, K=0..15;
// lanes 16-31 hold M=lane-16, K=16..31; VGPR q packs (K, K+1).
__device__ inline v16h gatherA(const float* buf, int stride, int kt, int lane) {
  int row = lane & 15;
  int hi  = lane >> 4;
  v16h a;
#pragma unroll
  for (int q = 0; q < 8; ++q) {
    int k = kt * 32 + hi * 16 + (q << 1);
    float2 f = *(const float2*)(buf + row * stride + k);
    a[2 * q]     = (_Float16)f.x;
    a[2 * q + 1] = (_Float16)f.y;
  }
  return a;
}

// Load a pre-packed B fragment (one 32B read per lane).
__device__ inline v16h loadB(const _Float16* pw, int Nt, int kt, int nt, int lane) {
  return *(const v16h*)(pw + ((((size_t)kt * Nt + nt) * 32 + lane) << 4));
}

// Async copy 16 bytes Global -> LDS (CDNA5 ASYNCcnt path, GV mode).
__device__ inline void async_g2l_b128(void* lds_dst, const void* gsrc) {
  unsigned dst = (unsigned)(unsigned long long)lds_dst;  // low 32 bits of the pointer = LDS address
  asm volatile("global_load_async_to_lds_b128 %0, %1, off"
               :: "v"(dst), "v"(gsrc) : "memory");
}
__device__ inline void async_wait0() {
  asm volatile("s_wait_asynccnt 0" ::: "memory");
}

// Pack weight W (K x N, f32 row-major) into per-lane WMMA B fragments (f16).
// K-contiguous B order (matches gatherA): lanes 0-15 -> col=lane, K=0..15 (VGPR q holds 2q,2q+1);
// lanes 16-31 -> col=lane-16, K=16..31.
__global__ __launch_bounds__(256) void pack_b16(const float* __restrict__ W,
                                                _Float16* __restrict__ dst,
                                                int K, int N) {
  int tid = blockIdx.x * blockDim.x + threadIdx.x;
  if (tid >= K * N) return;
  int e    = tid & 15;
  int lane = (tid >> 4) & 31;
  int rem  = tid >> 9;
  int Nt   = N >> 4;
  int nt   = rem % Nt;
  int kt   = rem / Nt;
  int col  = lane & 15;
  int hi   = lane >> 4;
  int k    = kt * 32 + hi * 16 + e;
  dst[tid] = (_Float16)W[(size_t)k * N + nt * 16 + col];
}

// ---------------- small linear kernels -----------------------------------

// Q/K/V = (X @ W + b) * node_mask   (1024 rows x 128)
__global__ __launch_bounds__(128) void qkv_kernel(
    const float* __restrict__ X, const float* __restrict__ nmask,
    const float* qw, const float* qb, const float* kw, const float* kb,
    const float* vw, const float* vb,
    float* __restrict__ Q, float* __restrict__ K, float* __restrict__ V) {
  __shared__ float xr[128];
  int row = blockIdx.x, c = threadIdx.x;
  float m = nmask[row];
  xr[c] = X[(size_t)row * 128 + c];
  __syncthreads();
  float aq = qb[c], ak = kb[c], av = vb[c];
  for (int k = 0; k < 128; ++k) {
    float x = xr[k];
    aq += x * qw[k * 128 + c];
    ak += x * kw[k * 128 + c];
    av += x * vw[k * 128 + c];
  }
  Q[(size_t)row * 128 + c] = aq * m;
  K[(size_t)row * 128 + c] = ak * m;
  V[(size_t)row * 128 + c] = av * m;
}

// Per-batch y-conditioning vectors: ye1, ye2, yx1, yx2, yy.
__global__ __launch_bounds__(128) void ysmall_kernel(
    const float* __restrict__ t,
    const float* yemw, const float* yemb,   // y_e_mul
    const float* yeaw, const float* yeab,   // y_e_add
    const float* yxmw, const float* yxmb,   // y_x_mul
    const float* yxaw, const float* yxab,   // y_x_add
    const float* yyw, const float* yyb,     // y_y
    float* ye1, float* ye2, float* yx1, float* yx2, float* yyv) {
  __shared__ float tr[128];
  int b = blockIdx.x, c = threadIdx.x;
  tr[c] = t[b * 128 + c];
  __syncthreads();
  float a1 = yeab[c], a2 = yemb[c], a3 = yxab[c], a4 = yxmb[c], a5 = yyb[c];
  for (int k = 0; k < 128; ++k) {
    float x = tr[k];
    a1 += x * yeaw[k * 128 + c];
    a2 += x * yemw[k * 128 + c];
    a3 += x * yxaw[k * 128 + c];
    a4 += x * yxmw[k * 128 + c];
    a5 += x * yyw[k * 128 + c];
  }
  ye1[b * 128 + c] = a1;
  ye2[b * 128 + c] = a2;
  yx1[b * 128 + c] = a3;
  yx2[b * 128 + c] = a4;
  yyv[b * 128 + c] = a5;
}

// ---------------- fused E mega-kernel (WMMA) -----------------------------
// Per 128-thread WG: 4 waves, each wave owns 16 pair-rows (fixed b,i; 16 j).
// Pipeline per wave tile:
//   E(16x64) --async--> LDS -> [WMMA] E1,E2(16x128) -> Y = QK/4*(E1+1)+E2
//   -> store Y (f16) -> ypre = ye1+(ye2+1)*Y -> [WMMA e_out] newE(16x64)
//   -> +E residual -> LN1 -> [WMMA mlp_e1] relu -> [WMMA mlp_e2]
//   -> +residual -> LN2 -> out E.
#define NW 4
__global__ __launch_bounds__(128) void e_mega_kernel(
    const float* __restrict__ E, const float* __restrict__ nmask,
    const float* __restrict__ Qws, const float* __restrict__ Kws,
    const float* __restrict__ ye1, const float* __restrict__ ye2,
    const _Float16* __restrict__ pw_emul, const float* __restrict__ b_emul,
    const _Float16* __restrict__ pw_eadd, const float* __restrict__ b_eadd,
    const _Float16* __restrict__ pw_eout, const float* __restrict__ b_eout,
    const _Float16* __restrict__ pw_m1,   const float* __restrict__ b_m1,
    const _Float16* __restrict__ pw_m2,   const float* __restrict__ b_m2,
    const float* __restrict__ g1, const float* __restrict__ bb1,
    const float* __restrict__ g2, const float* __restrict__ bb2,
    _Float16* __restrict__ Yws, float* __restrict__ outE) {
  __shared__ __align__(16) float buf64[NW][16 * 64];    // E tile / LN rows
  __shared__ __align__(16) float buf128[NW][16 * 128];  // ypre / hidden

  int w = threadIdx.x >> 5, lane = threadIdx.x & 31;
  int col = lane & 15, hi = lane >> 4;
  int bid = blockIdx.x;
  int jblk = bid & 3;
  int i    = (bid >> 2) & 255;
  int b    = bid >> 10;
  int j0   = jblk * 64 + w * 16;
  float* B64  = buf64[w];
  float* B128 = buf128[w];
  size_t pairBase = ((size_t)(b * 256 + i) * 256 + j0);

  // ---- stage 1: async-copy E tile (16 rows x 64 f32) Global -> LDS ----
#pragma unroll
  for (int tt = 0; tt < 8; ++tt) {
    int f = tt * 32 + lane;
    int row = f >> 4, c4 = f & 15;
    async_g2l_b128(B64 + row * 64 + c4 * 4,
                   E + (pairBase + row) * 64 + c4 * 4);
  }
  // overlap: mask loads while the async DMA runs
  float mi = nmask[b * 256 + i];
  float mj[8];
#pragma unroll
  for (int r = 0; r < 8; ++r) mj[r] = nmask[b * 256 + j0 + r + 8 * hi];
  async_wait0();
  __syncthreads();

  // ---- stage 2: e_mul / e_add GEMMs + Y + ypre ----
  for (int nt = 0; nt < 8; ++nt) {
    int ch = nt * 16 + col;
    float bm = b_emul[ch], ba = b_eadd[ch];
    v8f accM, accA;
#pragma unroll
    for (int r = 0; r < 8; ++r) { accM[r] = bm; accA[r] = ba; }
#pragma unroll
    for (int kt = 0; kt < 2; ++kt) {
      v16h a = gatherA(B64, 64, kt, lane);
      accM = wmma_f16(a, loadB(pw_emul, 8, kt, nt, lane), accM);
      accA = wmma_f16(a, loadB(pw_eadd, 8, kt, nt, lane), accA);
    }
    float qv = Qws[((size_t)b * 256 + i) * 128 + ch];
    float y1 = ye1[b * 128 + ch], y2 = ye2[b * 128 + ch];
#pragma unroll
    for (int r = 0; r < 8; ++r) {
      int m = r + 8 * hi;
      float mm = mi * mj[r];
      float e1 = accM[r] * mm, e2 = accA[r] * mm;
      float kv = Kws[((size_t)b * 256 + (j0 + m)) * 128 + ch];
      float y  = qv * kv * 0.25f;          // / sqrt(DF=16)
      y = y * (e1 + 1.0f) + e2;
      Yws[(pairBase + m) * 128 + ch] = (_Float16)y;
      B128[m * 128 + ch] = y1 + (y2 + 1.0f) * y;   // ypre
    }
  }
  __syncthreads();

  // ---- stage 3: e_out GEMM + mask + E residual ----
  for (int nt = 0; nt < 4; ++nt) {
    int c2 = nt * 16 + col;
    float bo = b_eout[c2];
    v8f acc;
#pragma unroll
    for (int r = 0; r < 8; ++r) acc[r] = bo;
#pragma unroll
    for (int kt = 0; kt < 4; ++kt) {
      v16h a = gatherA(B128, 128, kt, lane);
      acc = wmma_f16(a, loadB(pw_eout, 4, kt, nt, lane), acc);
    }
#pragma unroll
    for (int r = 0; r < 8; ++r) {
      int m = r + 8 * hi;
      float mm = mi * mj[r];
      B64[m * 64 + c2] = B64[m * 64 + c2] + acc[r] * mm;
    }
  }
  __syncthreads();

  // ---- stage 4: LayerNorm 1 (per row, 64 ch) ----
  if (lane < 16) {
    float* rowp = B64 + lane * 64;
    float s = 0.f, s2 = 0.f;
#pragma unroll
    for (int c = 0; c < 64; c += 4) {
      float4 v = *(float4*)(rowp + c);
      s  += v.x + v.y + v.z + v.w;
      s2 += v.x * v.x + v.y * v.y + v.z * v.z + v.w * v.w;
    }
    float mean = s * (1.f / 64.f);
    float var  = s2 * (1.f / 64.f) - mean * mean;
    float inv  = rsqrtf(var + EPS);
#pragma unroll
    for (int c = 0; c < 64; ++c)
      rowp[c] = (rowp[c] - mean) * inv * g1[c] + bb1[c];
  }
  __syncthreads();

  // ---- stage 5: mlp_e1 GEMM + relu ----
  for (int nt = 0; nt < 8; ++nt) {
    int ch = nt * 16 + col;
    float bh = b_m1[ch];
    v8f acc;
#pragma unroll
    for (int r = 0; r < 8; ++r) acc[r] = bh;
#pragma unroll
    for (int kt = 0; kt < 2; ++kt) {
      v16h a = gatherA(B64, 64, kt, lane);
      acc = wmma_f16(a, loadB(pw_m1, 8, kt, nt, lane), acc);
    }
#pragma unroll
    for (int r = 0; r < 8; ++r) {
      int m = r + 8 * hi;
      B128[m * 128 + ch] = fmaxf(acc[r], 0.f);
    }
  }
  __syncthreads();

  // ---- stage 6: mlp_e2 GEMM + residual ----
  for (int nt = 0; nt < 4; ++nt) {
    int c2 = nt * 16 + col;
    float bo = b_m2[c2];
    v8f acc;
#pragma unroll
    for (int r = 0; r < 8; ++r) acc[r] = bo;
#pragma unroll
    for (int kt = 0; kt < 4; ++kt) {
      v16h a = gatherA(B128, 128, kt, lane);
      acc = wmma_f16(a, loadB(pw_m2, 4, kt, nt, lane), acc);
    }
#pragma unroll
    for (int r = 0; r < 8; ++r) {
      int m = r + 8 * hi;
      B64[m * 64 + c2] = B64[m * 64 + c2] + acc[r];
    }
  }
  __syncthreads();

  // ---- stage 7: LayerNorm 2 + store final E ----
  if (lane < 16) {
    float* rowp = B64 + lane * 64;
    float s = 0.f, s2 = 0.f;
#pragma unroll
    for (int c = 0; c < 64; c += 4) {
      float4 v = *(float4*)(rowp + c);
      s  += v.x + v.y + v.z + v.w;
      s2 += v.x * v.x + v.y * v.y + v.z * v.z + v.w * v.w;
    }
    float mean = s * (1.f / 64.f);
    float var  = s2 * (1.f / 64.f) - mean * mean;
    float inv  = rsqrtf(var + EPS);
    size_t ob = (pairBase + lane) * 64;
#pragma unroll
    for (int c = 0; c < 64; c += 4) {
      float4 v = *(float4*)(rowp + c);
      v.x = (v.x - mean) * inv * g2[c]     + bb2[c];
      v.y = (v.y - mean) * inv * g2[c + 1] + bb2[c + 1];
      v.z = (v.z - mean) * inv * g2[c + 2] + bb2[c + 2];
      v.w = (v.w - mean) * inv * g2[c + 3] + bb2[c + 3];
      *(float4*)(outE + ob + c) = v;
    }
  }
}

// ---------------- online-softmax attention over j ------------------------
__global__ __launch_bounds__(128) void attn_kernel(
    const _Float16* __restrict__ Yws, const float* __restrict__ Vws,
    const float* __restrict__ nmask, float* __restrict__ wV) {
  int bi = blockIdx.x;        // b*256 + i
  int c  = threadIdx.x;       // channel
  int b  = bi >> 8;
  size_t base = (size_t)bi * 256 * 128;
  float m = -__builtin_inff(), s = 0.f, w = 0.f;
  for (int j = 0; j < 256; ++j) {
    if (j + 8 < 256)
      __builtin_prefetch(Yws + base + (size_t)(j + 8) * 128 + c, 0, 1);
    if (nmask[b * 256 + j] > 0.f) {
      float y  = (float)Yws[base + (size_t)j * 128 + c];
      float nm = fmaxf(m, y);
      float sc = expf(m - nm);   // first iter: exp(-inf)=0
      float e  = expf(y - nm);
      s = s * sc + e;
      w = w * sc + e * Vws[((size_t)b * 256 + j) * 128 + c];
      m = nm;
    }
  }
  wV[(size_t)bi * 128 + c] = w / s;
}

// ---------------- block reductions ---------------------------------------
__device__ inline float blk_sum128(float v, float* red) {
  int t = threadIdx.x;
  red[t] = v;
  __syncthreads();
  for (int s = 64; s > 0; s >>= 1) {
    if (t < s) red[t] += red[t + s];
    __syncthreads();
  }
  float r = red[0];
  __syncthreads();
  return r;
}

// ---------------- X path: x_out + LN1 + MLP_X + LN2 ----------------------
__global__ __launch_bounds__(128) void newx_kernel(
    const float* __restrict__ X, const float* __restrict__ nmask,
    const float* __restrict__ wV,
    const float* __restrict__ yx1, const float* __restrict__ yx2,
    const float* xow, const float* xob,
    const float* lng1, const float* lnb1,
    const float* m1w, const float* m1b, const float* m2w, const float* m2b,
    const float* lng2, const float* lnb2,
    float* __restrict__ outX) {
  __shared__ float arow[128], x1s[128], red[128];
  __shared__ float h[2048];
  int row = blockIdx.x, c = threadIdx.x, b = row >> 8;
  arow[c] = yx1[b * 128 + c] + (yx2[b * 128 + c] + 1.f) * wV[(size_t)row * 128 + c];
  __syncthreads();
  float nx = xob[c];
  for (int k = 0; k < 128; ++k) nx += arow[k] * xow[k * 128 + c];
  nx *= nmask[row];
  float x1 = X[(size_t)row * 128 + c] + nx;
  float s  = blk_sum128(x1, red);
  float s2 = blk_sum128(x1 * x1, red);
  float mean = s * (1.f / 128.f);
  float var  = s2 * (1.f / 128.f) - mean * mean;
  x1 = (x1 - mean) * rsqrtf(var + EPS) * lng1[c] + lnb1[c];
  x1s[c] = x1;
  __syncthreads();
  for (int o = c; o < 2048; o += 128) {
    float hv = m1b[o];
    for (int k = 0; k < 128; ++k) hv += x1s[k] * m1w[(size_t)k * 2048 + o];
    h[o] = fmaxf(hv, 0.f);
  }
  __syncthreads();
  float x2 = m2b[c];
  for (int k = 0; k < 2048; ++k) x2 += h[k] * m2w[(size_t)k * 128 + c];
  x2 += x1s[c];
  s  = blk_sum128(x2, red);
  s2 = blk_sum128(x2 * x2, red);
  mean = s * (1.f / 128.f);
  var  = s2 * (1.f / 128.f) - mean * mean;
  outX[(size_t)row * 128 + c] = (x2 - mean) * rsqrtf(var + EPS) * lng2[c] + lnb2[c];
}

// ---------------- pooled stats of X / E ----------------------------------
__global__ __launch_bounds__(128) void xstats_kernel(const float* __restrict__ X,
                                                     float* __restrict__ xs) {
  int b = blockIdx.x, c = threadIdx.x;
  float s = 0.f, s2 = 0.f, mn = __builtin_inff(), mx = -__builtin_inff();
  for (int r = 0; r < 256; ++r) {
    float v = X[((size_t)b * 256 + r) * 128 + c];
    s += v; s2 += v * v;
    mn = fminf(mn, v); mx = fmaxf(mx, v);
  }
  xs[b * 512 + c]       = s * (1.f / 256.f);
  xs[b * 512 + 128 + c] = mn;
  xs[b * 512 + 256 + c] = mx;
  xs[b * 512 + 384 + c] = sqrtf(fmaxf(0.f, (s2 - s * s * (1.f / 256.f)) * (1.f / 255.f)));
}

__global__ __launch_bounds__(64) void estats1_kernel(const float* __restrict__ E,
                                                     float* __restrict__ epart) {
  int blk = blockIdx.x, ch = threadIdx.x;
  int b = blk >> 6, slice = blk & 63;
  size_t base = ((size_t)b * 65536 + (size_t)slice * 1024) * 64;
  float s = 0.f, s2 = 0.f, mn = __builtin_inff(), mx = -__builtin_inff();
  for (int r = 0; r < 1024; ++r) {
    float v = E[base + (size_t)r * 64 + ch];
    s += v; s2 += v * v;
    mn = fminf(mn, v); mx = fmaxf(mx, v);
  }
  size_t o = (size_t)(b * 64 + slice) * 4 * 64 + ch;
  epart[o] = s; epart[o + 64] = s2; epart[o + 128] = mn; epart[o + 192] = mx;
}

__global__ __launch_bounds__(64) void estats2_kernel(const float* __restrict__ epart,
                                                     float* __restrict__ es) {
  int b = blockIdx.x, ch = threadIdx.x;
  float s = 0.f, s2 = 0.f, mn = __builtin_inff(), mx = -__builtin_inff();
  for (int sl = 0; sl < 64; ++sl) {
    size_t o = (size_t)(b * 64 + sl) * 4 * 64 + ch;
    s += epart[o]; s2 += epart[o + 64];
    mn = fminf(mn, epart[o + 128]); mx = fmaxf(mx, epart[o + 192]);
  }
  const float N = 65536.f;
  es[b * 256 + ch]        = s / N;
  es[b * 256 + 64 + ch]   = mn;
  es[b * 256 + 128 + ch]  = mx;
  es[b * 256 + 192 + ch]  = sqrtf(fmaxf(0.f, (s2 - s * s / N) / (N - 1.f)));
}

// ---------------- y path final -------------------------------------------
__global__ __launch_bounds__(128) void yfinal_kernel(
    const float* __restrict__ t, const float* __restrict__ yyv,
    const float* __restrict__ xs, const float* __restrict__ es,
    const float* xyw, const float* xyb, const float* eyw, const float* eyb,
    const float* o1w, const float* o1b, const float* o2w, const float* o2b,
    const float* lng1, const float* lnb1,
    const float* m1w, const float* m1b, const float* m2w, const float* m2b,
    const float* lng2, const float* lnb2,
    float* __restrict__ outT) {
  __shared__ float sA[128], sB[128], red[128];
  __shared__ float h[2048];
  int b = blockIdx.x, c = threadIdx.x;
  float v = yyv[b * 128 + c] + xyb[c] + eyb[c];
  for (int k = 0; k < 512; ++k) v += xs[b * 512 + k] * xyw[(size_t)k * 128 + c];
  for (int k = 0; k < 256; ++k) v += es[b * 256 + k] * eyw[(size_t)k * 128 + c];
  sA[c] = v;
  __syncthreads();
  float h1 = o1b[c];
  for (int k = 0; k < 128; ++k) h1 += sA[k] * o1w[k * 128 + c];
  h1 = fmaxf(h1, 0.f);
  sB[c] = h1;
  __syncthreads();
  float ny = o2b[c];
  for (int k = 0; k < 128; ++k) ny += sB[k] * o2w[k * 128 + c];
  float t1 = t[b * 128 + c] + ny;
  float s  = blk_sum128(t1, red);
  float s2 = blk_sum128(t1 * t1, red);
  float mean = s * (1.f / 128.f);
  float var  = s2 * (1.f / 128.f) - mean * mean;
  t1 = (t1 - mean) * rsqrtf(var + EPS) * lng1[c] + lnb1[c];
  sA[c] = t1;
  __syncthreads();
  for (int o = c; o < 2048; o += 128) {
    float hv = m1b[o];
    for (int k = 0; k < 128; ++k) hv += sA[k] * m1w[(size_t)k * 2048 + o];
    h[o] = fmaxf(hv, 0.f);
  }
  __syncthreads();
  float t2 = m2b[c];
  for (int k = 0; k < 2048; ++k) t2 += h[k] * m2w[(size_t)k * 128 + c];
  t2 += sA[c];
  s  = blk_sum128(t2, red);
  s2 = blk_sum128(t2 * t2, red);
  mean = s * (1.f / 128.f);
  var  = s2 * (1.f / 128.f) - mean * mean;
  outT[b * 128 + c] = (t2 - mean) * rsqrtf(var + EPS) * lng2[c] + lnb2[c];
}

// ---------------- host-side launch ---------------------------------------

extern "C" void kernel_launch(void* const* d_in, const int* in_sizes, int n_in,
                              void* d_out, int out_size, void* d_ws, size_t ws_size,
                              hipStream_t stream) {
  const float* X     = (const float*)d_in[0];
  const float* E     = (const float*)d_in[1];
  const float* t     = (const float*)d_in[2];
  const float* nmask = (const float*)d_in[3];
#define P(i) ((const float*)d_in[i])
  // params: 4 q_w,5 q_b,6 k_w,7 k_b,8 v_w,9 v_b,10 e_add_w,11 e_add_b,
  // 12 e_mul_w,13 e_mul_b,14 y_e_mul_w,15 b,16 y_e_add_w,17 b,18 y_x_mul_w,19 b,
  // 20 y_x_add_w,21 b,22 y_y_w,23 b,24 x_y_w,25 b,26 e_y_w,27 b,28 x_out_w,29 b,
  // 30 e_out_w,31 b,32 y_out1_w,33 b,34 y_out2_w,35 b,36 mlp_x1_w,37 b,
  // 38 mlp_x2_w,39 b,40 mlp_e1_w,41 b,42 mlp_e2_w,43 b,44 mlp_y1_w,45 b,
  // 46 mlp_y2_w,47 b,48 ln_x1_g,49 b,50 ln_e1_g,51 b,52 ln_y1_g,53 b,
  // 54 ln_x2_g,55 b,56 ln_e2_g,57 b,58 ln_y2_g,59 b

  float* wsf = (float*)d_ws;
  float* Q   = wsf;                // 131072
  float* Kw  = wsf + 131072;       // 131072
  float* V   = wsf + 262144;       // 131072
  float* wV  = wsf + 393216;       // 131072
  float* ye1 = wsf + 524288;       // 512
  float* ye2 = ye1 + 512;
  float* yx1 = ye2 + 512;
  float* yx2 = yx1 + 512;
  float* yyv = yx2 + 512;
  float* xs  = yyv + 512;          // 2048
  float* es  = xs + 2048;          // 1024
  float* epart = es + 1024;        // 65536
  _Float16* pw      = (_Float16*)(wsf + 595456);   // 32B-aligned
  _Float16* pw_emul = pw;
  _Float16* pw_eadd = pw + 8192;
  _Float16* pw_eout = pw + 16384;
  _Float16* pw_m1   = pw + 24576;
  _Float16* pw_m2   = pw + 32768;
  _Float16* Yws     = pw + 40960;  // 4*256*256*128 f16 = 64 MiB

  float* outX = (float*)d_out;
  float* outE = outX + 131072;
  float* outT = outE + 16777216;

  // pack the 5 E-path weights into WMMA B-fragment layout (f16)
  pack_b16<<<32, 256, 0, stream>>>(P(12), pw_emul, 64, 128);   // e_mul_w
  pack_b16<<<32, 256, 0, stream>>>(P(10), pw_eadd, 64, 128);   // e_add_w
  pack_b16<<<32, 256, 0, stream>>>(P(30), pw_eout, 128, 64);   // e_out_w
  pack_b16<<<32, 256, 0, stream>>>(P(40), pw_m1, 64, 128);     // mlp_e1_w
  pack_b16<<<32, 256, 0, stream>>>(P(42), pw_m2, 128, 64);     // mlp_e2_w

  qkv_kernel<<<1024, 128, 0, stream>>>(X, nmask, P(4), P(5), P(6), P(7), P(8), P(9),
                                       Q, Kw, V);
  ysmall_kernel<<<4, 128, 0, stream>>>(t, P(14), P(15), P(16), P(17), P(18), P(19),
                                       P(20), P(21), P(22), P(23),
                                       ye1, ye2, yx1, yx2, yyv);

  e_mega_kernel<<<4096, 128, 0, stream>>>(
      E, nmask, Q, Kw, ye1, ye2,
      pw_emul, P(13), pw_eadd, P(11), pw_eout, P(31),
      pw_m1, P(41), pw_m2, P(43),
      P(50), P(51), P(56), P(57), Yws, outE);

  attn_kernel<<<1024, 128, 0, stream>>>(Yws, V, nmask, wV);
  newx_kernel<<<1024, 128, 0, stream>>>(X, nmask, wV, yx1, yx2,
                                        P(28), P(29), P(48), P(49),
                                        P(36), P(37), P(38), P(39),
                                        P(54), P(55), outX);

  xstats_kernel<<<4, 128, 0, stream>>>(X, xs);
  estats1_kernel<<<256, 64, 0, stream>>>(E, epart);
  estats2_kernel<<<4, 64, 0, stream>>>(epart, es);
  yfinal_kernel<<<4, 128, 0, stream>>>(t, yyv, xs, es,
                                       P(24), P(25), P(26), P(27),
                                       P(32), P(33), P(34), P(35),
                                       P(52), P(53),
                                       P(44), P(45), P(46), P(47),
                                       P(58), P(59), outT);
#undef P
}
